// CrossAttnPoolingNet_51204600103340
// MI455X (gfx1250) — compile-verified, matrix operand order checked by reading, not run on hardware
//
#include <hip/hip_runtime.h>
#include <hip/hip_bf16.h>

// MI455X (gfx1250) fused CrossAttnPoolingNet.
//  - v_wmma_f32_16x16x32_bf16 for the H GEMM and the MLP GEMM
//  - global_load_async_to_lds_b128 + s_wait_asynccnt for double-buffered staging
//  - K/V projections folded through the single query vector
// Workspace use: ~3.2 MB.

typedef __attribute__((ext_vector_type(16))) __bf16 v16bf;
typedef __attribute__((ext_vector_type(8)))  __bf16 v8bf;
typedef __attribute__((ext_vector_type(8)))  float  v8f;
typedef __attribute__((ext_vector_type(2)))  __bf16 bf16x2;

#define B_   2048
#define T_   200
#define TP_  208      // T padded to 13*16
#define D_   256
#define M_   128
#define FD_  384
#define NEGV (-1.0e9f)

__device__ __forceinline__ int kb_pat(int j, int half) {
  // 16-bit WMMA K-pair interleave: VGPR0..3 -> K {0,2,4,6}+8*half, VGPR4..7 -> +16
  return (j < 4 ? 2 * j : 16 + 2 * (j - 4)) + 8 * half;
}

__device__ __forceinline__ v8f wmma_bf16(v16bf a, v16bf b, v8f c) {
  return __builtin_amdgcn_wmma_f32_16x16x32_bf16(false, a, false, b, (short)0, c,
                                                 false, false);
}

__device__ __forceinline__ unsigned lds_off(const void* p) {
  return (unsigned)(size_t)p;   // low 32 bits of flat shared address = LDS offset
}

// ---------------------------------------------------------------------------
// Prep 1: bf16 weight repack (paired-K B layout) + positional-encoding table.
//   wpbP : [k/2][m][2]  (256x128)     fc1P : [k/2][n][2]  (384x512)
// ---------------------------------------------------------------------------
__global__ __launch_bounds__(256) void prep_convert(
    const float* __restrict__ Wproj, const float* __restrict__ fc1w,
    __bf16* __restrict__ wpbP, __bf16* __restrict__ fc1P,
    float* __restrict__ pe) {
  int idx = blockIdx.x * 256 + threadIdx.x;
  if (idx < 256 * 128) {
    int k = idx >> 7, m = idx & 127;
    wpbP[(k >> 1) * 256 + m * 2 + (k & 1)] = (__bf16)Wproj[m * 256 + k];
  } else if (idx < 256 * 128 + 384 * 512) {
    int i2 = idx - 256 * 128;
    int k = i2 >> 9, n = i2 & 511;
    fc1P[(k >> 1) * 1024 + n * 2 + (k & 1)] = (__bf16)fc1w[n * 384 + k];
  } else {
    int i3 = idx - (256 * 128 + 384 * 512);   // posenc [208][128], zero padded
    int t = i3 >> 7, m = i3 & 127;
    float v = 0.f;
    if (t < T_) {
      int i = m >> 1;
      float div = __expf(-9.210340372f * (float)(2 * i) / 128.f); // ln(1e4)
      float a = (float)t * div;
      v = (m & 1) ? __sinf(a) : __cosf(a);
    }
    pe[i3] = v;
  }
}

// ---------------------------------------------------------------------------
// Prep 2: qh = query@Wq^T + bq; kq[h] = Wk_h^T qh_h; beta[h] = qh_h . bk_h
// ---------------------------------------------------------------------------
__global__ __launch_bounds__(256) void prep_qk(
    const float* __restrict__ query, const float* __restrict__ ipw,
    const float* __restrict__ ipb, float* __restrict__ kq,
    float* __restrict__ beta) {
  __shared__ float qh[128];
  int tid = threadIdx.x;
  if (tid < 128) {
    float s = ipb[tid];
    for (int m = 0; m < 128; ++m) s += query[m] * ipw[tid * 128 + m];
    qh[tid] = s;
  }
  __syncthreads();
  {
    int h = tid >> 7, m = tid & 127;
    float s = 0.f;
    for (int d = 0; d < 64; ++d)
      s += qh[h * 64 + d] * ipw[(128 + h * 64 + d) * 128 + m];
    kq[tid] = s;
  }
  if (tid < 2) {
    float s = 0.f;
    for (int d = 0; d < 64; ++d) s += qh[tid * 64 + d] * ipb[128 + tid * 64 + d];
    beta[tid] = s;
  }
}

// ---------------------------------------------------------------------------
// Main fused attention kernel: one workgroup (8 waves) per batch row.
// Dynamic LDS layout (bytes):
//   xpf  f32 [16][264]  @ 0       (async landing buffer, 16896)
//   xpb  bf16[16][264]  @ 16896   (8448)
//   Hl   bf16[208][132] @ 25344   (54912)
//   sc   f32 [2][208]   @ 80256   (1664)
//   red  f32 [2], flag int        (12)
// xpf is re-aliased for kq/u/ctx after phase 1.
// ---------------------------------------------------------------------------
__global__ __launch_bounds__(256) void attn_kernel(
    const float* __restrict__ xpast, const int* __restrict__ maskp,
    const float* __restrict__ pe, const __bf16* __restrict__ wpbP,
    const float* __restrict__ kq, const float* __restrict__ beta,
    const float* __restrict__ ipw, const float* __restrict__ ipb,
    const float* __restrict__ outw, const float* __restrict__ outb,
    float* __restrict__ zws, float* __restrict__ out) {
  extern __shared__ char smem[];
  float*  xpf = (float*)smem;                    // [16][264]
  __bf16* xpb = (__bf16*)(smem + 16896);         // [16][264]
  __bf16* Hl  = (__bf16*)(smem + 25344);         // [208][132]
  float*  sc  = (float*)(smem + 80256);          // [2][208]
  float*  red = (float*)(smem + 81920);          // [2]
  int*    flag = (int*)(smem + 81928);

  const int tid  = threadIdx.x;
  const int wave = tid >> 5;
  const int lane = tid & 31;
  const int ln   = lane & 15;
  const int half = lane >> 4;
  const int b    = blockIdx.x;

  float* kql  = xpf;          // aliases, live after phase 1 only
  float* ul   = xpf + 256;
  float* ctxl = xpf + 512;

  // -- resident B tiles: W_projB strip for this wave's 16 output columns -----
  v16bf Bt[8];
  {
    const int mcol = wave * 16 + ln;
#pragma unroll
    for (int ks = 0; ks < 8; ++ks)
#pragma unroll
      for (int j = 0; j < 8; ++j) {
        int k = ks * 32 + kb_pat(j, half);
        bf16x2 p = *(const bf16x2*)&wpbP[(k >> 1) * 256 + mcol * 2];
        Bt[ks][2 * j]     = p[0];
        Bt[ks][2 * j + 1] = p[1];
      }
  }

  // -- phase 1: H = x_past @ W_proj^T + posenc, 13 chunks of 16 time rows ----
  const int r16 = tid >> 4;
  const int kc  = (tid & 15) * 16;
  auto issue_async = [&](int c) {     // async-copy one 16x256 f32 chunk to LDS
    int t = c * 16 + r16;
    if (t < T_) {
      unsigned dst  = lds_off(xpf + r16 * 264 + kc);
      unsigned goff = (unsigned)((((size_t)b * T_ + t) * D_ + kc) * 4);
      asm volatile(
          "global_load_async_to_lds_b128 %0, %1, %2 offset:0\n\t"
          "global_load_async_to_lds_b128 %0, %1, %2 offset:16\n\t"
          "global_load_async_to_lds_b128 %0, %1, %2 offset:32\n\t"
          "global_load_async_to_lds_b128 %0, %1, %2 offset:48"
          :: "v"(dst), "v"(goff), "s"(xpast) : "memory");
    }
  };

  issue_async(0);
  asm volatile("s_wait_asynccnt 0" ::: "memory");
  __syncthreads();

  for (int c = 0; c < 13; ++c) {
    // convert chunk c (LDS f32 -> regs)
    float fx[16];
    {
      int t = c * 16 + r16;
      if (t < T_) {
#pragma unroll
        for (int i = 0; i < 16; ++i) fx[i] = xpf[r16 * 264 + kc + i];
      } else {
#pragma unroll
        for (int i = 0; i < 16; ++i) fx[i] = 0.f;
      }
    }
    __syncthreads();                      // xpf reads done
#pragma unroll
    for (int i = 0; i < 16; ++i) xpb[r16 * 264 + kc + i] = (__bf16)fx[i];
    if (c + 1 < 13) issue_async(c + 1);   // overlap next copy with WMMA
    __syncthreads();                      // xpb visible

    // A tiles into distinct registers, then back-to-back WMMA chain
    v16bf at[8];
#pragma unroll
    for (int ks = 0; ks < 8; ++ks)
#pragma unroll
      for (int j = 0; j < 8; ++j) {
        int k = ks * 32 + kb_pat(j, half);
        bf16x2 p = *(const bf16x2*)&xpb[ln * 264 + k];
        at[ks][2 * j]     = p[0];
        at[ks][2 * j + 1] = p[1];
      }
    v8f acc = {0.f, 0.f, 0.f, 0.f, 0.f, 0.f, 0.f, 0.f};
#pragma unroll
    for (int ks = 0; ks < 8; ++ks) acc = wmma_bf16(at[ks], Bt[ks], acc);

    const int col = wave * 16 + ln;
#pragma unroll
    for (int r = 0; r < 8; ++r) {
      int trow = c * 16 + r + 8 * half;
      float v = acc[r] + pe[trow * 128 + col];
      Hl[trow * 132 + col] = (__bf16)v;
    }
    asm volatile("s_wait_asynccnt 0" ::: "memory");
    __syncthreads();                      // next chunk landed in xpf
  }

  // -- stage kq into LDS (xpf now dead), reset has_past flag -----------------
  kql[tid] = kq[tid];
  if (tid == 0) *flag = 0;
  __syncthreads();

  // -- phase 2: masked scores s[h,t] = 0.125*(H[t].kq[h] + beta[h]) ----------
  for (int idx = tid; idx < 2 * TP_; idx += 256) {
    int h = idx / TP_;
    int t = idx % TP_;
    bool valid = (t < T_) && (maskp[b * T_ + t] != 0);
    float s = NEGV;
    if (valid) {
      float d = 0.f;
      for (int i = 0; i < 128; ++i) d += kql[h * 128 + i] * (float)Hl[t * 132 + i];
      s = 0.125f * (d + beta[h]);
      atomicOr(flag, 1);
    }
    sc[h * TP_ + t] = s;
  }
  __syncthreads();

  // -- phase 3: softmax per head (wave-shuffle reductions, wave32) -----------
  if (wave < 2) {
    int h = wave;
    float mx = NEGV;
    for (int t = lane; t < TP_; t += 32) mx = fmaxf(mx, sc[h * TP_ + t]);
#pragma unroll
    for (int m = 16; m >= 1; m >>= 1) mx = fmaxf(mx, __shfl_xor(mx, m, 32));
    float sm = 0.f;
    for (int t = lane; t < TP_; t += 32) {
      float e = __expf(sc[h * TP_ + t] - mx);
      sc[h * TP_ + t] = e;
      sm += e;
    }
#pragma unroll
    for (int m = 16; m >= 1; m >>= 1) sm += __shfl_xor(sm, m, 32);
    if (lane == 0) red[h] = 1.f / sm;
  }
  __syncthreads();
  for (int idx = tid; idx < 2 * TP_; idx += 256) {
    int h = idx / TP_, t = idx % TP_;
    sc[h * TP_ + t] *= red[h];
  }
  __syncthreads();
  const int hp = *flag;
  if (tid < T_) {                              // full_attn output
    float fa = hp ? 0.5f * (sc[tid] + sc[TP_ + tid]) : 0.f;
    out[B_ + (size_t)b * T_ + tid] = fa;
  }

  // -- phase 4: u[h] = sum_t attn[h,t]*H[t]  (folded V projection) -----------
  {
    int h = tid >> 7, cm = tid & 127;
    float acc = 0.f;
    for (int t = 0; t < TP_; ++t) acc += sc[h * TP_ + t] * (float)Hl[t * 132 + cm];
    ul[tid] = acc;
  }
  __syncthreads();
  if (tid < 128) {                             // ctx = Wv_h u[h] + bv
    int h = tid >> 6;
    float c = ipb[2 * M_ + tid];
    for (int m = 0; m < 128; ++m)
      c += ipw[(2 * M_ + tid) * M_ + m] * ul[h * 128 + m];
    ctxl[tid] = c;
  }
  __syncthreads();
  if (tid < 128) {                             // z = ctx @ out_w^T + out_b
    float z = outb[tid];
    for (int m = 0; m < 128; ++m) z += outw[tid * 128 + m] * ctxl[m];
    zws[(size_t)b * 128 + tid] = hp ? z : 0.f;
  }
}

// ---------------------------------------------------------------------------
// feats = [z | x_current] packed to bf16 in global (A matrix for the MLP GEMM)
// ---------------------------------------------------------------------------
__global__ __launch_bounds__(256) void feats_pack(
    const float* __restrict__ zws, const float* __restrict__ xcur,
    __bf16* __restrict__ featsb) {
  int idx = blockIdx.x * 256 + threadIdx.x;    // 2048*384
  int r = idx / FD_, c = idx % FD_;
  float v = (c < 128) ? zws[(size_t)r * 128 + c]
                      : xcur[(size_t)r * 256 + (c - 128)];
  featsb[idx] = (__bf16)v;
}

// ---------------------------------------------------------------------------
// MLP: (2048x384)@(384x512) -> relu -> fc2, WMMA bf16.
// 64 WGs x 32 rows. fc1 weights double-buffered into LDS with async copies.
// Dynamic LDS: fcb bf16[2][16][1032] @0 (66048 B) | lg f32[32] @66048
// ---------------------------------------------------------------------------
__global__ __launch_bounds__(256) void mlp_kernel(
    const __bf16* __restrict__ featsb, const __bf16* __restrict__ fc1P,
    const float* __restrict__ fc1bias, const float* __restrict__ fc2w,
    const float* __restrict__ fc2b, float* __restrict__ out) {
  extern __shared__ char smem[];
  __bf16* fcb = (__bf16*)smem;                 // [2][16][1032]
  float*  lg  = (float*)(smem + 66048);        // [32]

  const int tid = threadIdx.x, wave = tid >> 5, lane = tid & 31;
  const int ln = lane & 15, half = lane >> 4;
  const int rb = blockIdx.x * 32;
  if (tid < 32) lg[tid] = 0.f;

  const int j2 = tid >> 4, seg = tid & 15;     // 16 pair-rows x 16 segments
  auto issue_slice = [&](int ks, int buf) {    // 32 K-rows (16 pair-rows) x 512
    unsigned dst  = lds_off(fcb + ((size_t)buf * 16 + j2) * 1032 + seg * 64);
    unsigned goff = (unsigned)((((size_t)ks * 16 + j2) * 1024 + seg * 64) * 2);
    asm volatile(
        "global_load_async_to_lds_b128 %0, %1, %2 offset:0\n\t"
        "global_load_async_to_lds_b128 %0, %1, %2 offset:16\n\t"
        "global_load_async_to_lds_b128 %0, %1, %2 offset:32\n\t"
        "global_load_async_to_lds_b128 %0, %1, %2 offset:48\n\t"
        "global_load_async_to_lds_b128 %0, %1, %2 offset:64\n\t"
        "global_load_async_to_lds_b128 %0, %1, %2 offset:80\n\t"
        "global_load_async_to_lds_b128 %0, %1, %2 offset:96\n\t"
        "global_load_async_to_lds_b128 %0, %1, %2 offset:112"
        :: "v"(dst), "v"(goff), "s"(fc1P) : "memory");
  };

  issue_slice(0, 0);
  const int rt = wave >> 2;                    // 16-row tile
  const int cb = (wave & 3) * 128;             // 128-col strip
  const __bf16* arow = featsb + (size_t)(rb + rt * 16 + ln) * FD_;
  v8f acc[8];
  v8f zero = {0.f, 0.f, 0.f, 0.f, 0.f, 0.f, 0.f, 0.f};
#pragma unroll
  for (int i = 0; i < 8; ++i) acc[i] = zero;

  asm volatile("s_wait_asynccnt 0" ::: "memory");
  __syncthreads();
  int buf = 0;
  for (int ks = 0; ks < 12; ++ks) {
    if (ks + 1 < 12) issue_slice(ks + 1, buf ^ 1);
    // A tile: two aligned b128 loads from packed feats
    v16bf a;
    v8bf lo = *(const v8bf*)(arow + ks * 32 + 8 * half);
    v8bf hi = *(const v8bf*)(arow + ks * 32 + 16 + 8 * half);
#pragma unroll
    for (int j = 0; j < 8; ++j) { a[j] = lo[j]; a[8 + j] = hi[j]; }

    const __bf16* bbase = fcb + (size_t)buf * 16 * 1032;
#pragma unroll
    for (int ct = 0; ct < 8; ++ct) {
      int col = cb + ct * 16 + ln;
      v16bf bb;
#pragma unroll
      for (int j = 0; j < 8; ++j) {
        int j2l = kb_pat(j, half) >> 1;
        bf16x2 p = *(const bf16x2*)(bbase + j2l * 1032 + col * 2);
        bb[2 * j]     = p[0];
        bb[2 * j + 1] = p[1];
      }
      acc[ct] = wmma_bf16(a, bb, acc[ct]);
    }
    asm volatile("s_wait_asynccnt 0" ::: "memory");
    __syncthreads();
    buf ^= 1;
  }

  // bias + relu + fc2 partial reduction straight from accumulators
#pragma unroll
  for (int r = 0; r < 8; ++r) {
    int rowl = rt * 16 + r + 8 * half;
    float part = 0.f;
#pragma unroll
    for (int ct = 0; ct < 8; ++ct) {
      int col = cb + ct * 16 + ln;
      float v = acc[ct][r] + fc1bias[col];
      v = v > 0.f ? v : 0.f;
      part += v * fc2w[col];
    }
    atomicAdd(&lg[rowl], part);
  }
  __syncthreads();
  if (tid < 32) out[rb + tid] = lg[tid] + fc2b[0];
}

// ---------------------------------------------------------------------------
extern "C" void kernel_launch(void* const* d_in, const int* in_sizes, int n_in,
                              void* d_out, int out_size, void* d_ws,
                              size_t ws_size, hipStream_t stream) {
  const float* xcur    = (const float*)d_in[0];
  const float* xpast   = (const float*)d_in[1];
  const int*   maskp   = (const int*)d_in[2];
  const float* Wproj   = (const float*)d_in[3];
  const float* query   = (const float*)d_in[4];
  const float* ipw     = (const float*)d_in[5];
  const float* ipb     = (const float*)d_in[6];
  const float* outw    = (const float*)d_in[7];
  const float* outb    = (const float*)d_in[8];
  const float* fc1w    = (const float*)d_in[9];
  const float* fc1bias = (const float*)d_in[10];
  const float* fc2w    = (const float*)d_in[11];
  const float* fc2b    = (const float*)d_in[12];
  float* out = (float*)d_out;

  // workspace layout (~3.2 MB)
  char* ws = (char*)d_ws;
  __bf16* wpbP   = (__bf16*)(ws);                          // 256*128 bf16
  __bf16* fc1P   = (__bf16*)(ws + 65536);                  // 384*512 bf16
  float*  pe     = (float*)(ws + 65536 + 393216);          // 208*128 f32
  float*  kq     = (float*)(ws + 65536 + 393216 + 106496); // 256 f32
  float*  beta   = kq + 256;                               // 2 f32
  float*  zws    = beta + 64;                              // 2048*128 f32
  __bf16* featsb = (__bf16*)((char*)(zws + 2048 * 128));   // 2048*384 bf16

  prep_convert<<<1000, 256, 0, stream>>>(Wproj, fc1w, wpbP, fc1P, pe);
  prep_qk<<<1, 256, 0, stream>>>(query, ipw, ipb, kq, beta);
  attn_kernel<<<B_, 256, 81952, stream>>>(xpast, maskp, pe, wpbP, kq, beta, ipw,
                                          ipb, outw, outb, zws, out);
  feats_pack<<<B_ * FD_ / 256, 256, 0, stream>>>(zws, xcur, featsb);
  mlp_kernel<<<B_ / 32, 256, 66304, stream>>>(featsb, fc1P, fc1bias, fc2w, fc2b,
                                              out);
}
